// layerATT_52338471469741
// MI455X (gfx1250) — matrix operand order repaired and checked by reading, not compiled, not run on hardware
//
#include <hip/hip_runtime.h>
#include <hip/hip_bf16.h>

// ---------------------------------------------------------------------------
// Shapes (from reference)
// ---------------------------------------------------------------------------
#define BB   32
#define CHN  1024
#define C4N  256
#define PP   576      // H*W = EDIM
#define SS   256      // attention sequence = C4
#define EE   576      // attention embed = EDIM
#define NHD  9
#define HD   64
#define LL   576      // mamba sequence
#define DIN  512
#define DSN  64
#define RN   16

typedef __bf16 bf16_t;
typedef __attribute__((ext_vector_type(16))) __bf16 v16bf;
typedef __attribute__((ext_vector_type(8)))  float  v8f;

#if __has_builtin(__builtin_amdgcn_global_load_async_to_lds_b128)
#define HAVE_ASYNC_LDS 1
typedef int v4i_async __attribute__((vector_size(16)));
typedef __attribute__((address_space(1))) v4i_async* g_v4i_ptr;
typedef __attribute__((address_space(3))) v4i_async* l_v4i_ptr;
#endif

union FragU { v16bf v; unsigned u[8]; };

__device__ __forceinline__ unsigned pack2(float a, float b) {
    union { bf16_t h[2]; unsigned u; } p;
    p.h[0] = (bf16_t)a; p.h[1] = (bf16_t)b;
    return p.u;
}

// ---------------------------------------------------------------------------
// Generic batched WMMA GEMM:
//   Out[bz](MxN) = act( scaleA * A[bz](MxK) * B[bz](KxN) + bias ) + Res
// bz = bo*nInner + bi ; pointer offset = bo*stride_o + bi*stride_i.
// B element: transB ? Bp[n*ldb+k] : Bp[k*ldb+n].
// bias: 0 none, 1 over N, 2 over M. act: 0 none, 1 gelu(exact), 2 softplus.
// Residual for (m,n): Res[m*ldrM + n*ldrN].
// Block tile 128x64, 8 waves; each wave computes 32x32 via 2x2 WMMA tiles
// (4 x v_wmma_f32_16x16x32_bf16 per K-step). A tile k-major, B tile n-major
// in LDS so fragments assemble from aligned 32-bit (merged to b128) LDS reads.
// Interior blocks run a software-pipelined path: store tile k regs->LDS,
// issue tile k+1 global_load_b128s, then compute tile k (loads overlap WMMA).
// ---------------------------------------------------------------------------
__global__ __launch_bounds__(256)
void wmma_gemm_kernel(
    const float* __restrict__ A,  long long sAo, long long sAi, int lda, float scaleA,
    const float* __restrict__ Bp, long long sBo, long long sBi, int ldb, int transB,
    float* __restrict__ C,        long long sCo, long long sCi, int ldc,
    const float* __restrict__ bias, int biasMode,
    const float* __restrict__ Res,  long long sRo, long long sRi,
    long long ldrM, long long ldrN,
    int M, int N, int Kdim, int nInner, int act)
{
    __shared__ __align__(16) bf16_t As [128][40];  // 128 x 32 (+8 pad), k-major
    __shared__ __align__(16) bf16_t BsT[ 64][40];  // 64 x 32 (+8 pad), n-major [n][k]

    const int bz = blockIdx.z;
    const int bo = bz / nInner, bi = bz % nInner;
    A  += (long long)bo * sAo + (long long)bi * sAi;
    Bp += (long long)bo * sBo + (long long)bi * sBi;
    C  += (long long)bo * sCo + (long long)bi * sCi;
    if (Res) Res += (long long)bo * sRo + (long long)bi * sRi;

    const int m0 = blockIdx.y * 128;
    const int n0 = blockIdx.x * 64;
    const int tid  = threadIdx.x;
    const int wave = tid >> 5;
    const int lane = tid & 31;
    const int wm = (wave >> 1) * 32;   // 0,32,64,96
    const int wn = (wave & 1) * 32;    // 0,32
    const int half = lane >> 4;
    const int mr = lane & 15;
    const int nc = lane & 15;

    v8f acc00 = {0.f,0.f,0.f,0.f,0.f,0.f,0.f,0.f};
    v8f acc01 = acc00, acc10 = acc00, acc11 = acc00;

    // Fragment assembly + 4 WMMAs on the currently staged LDS tiles.
    auto computeStep = [&]() {
        const bf16_t* a0p = &As [wm +      mr][0];
        const bf16_t* a1p = &As [wm + 16 + mr][0];
        const bf16_t* b0p = &BsT[wn +      nc][0];
        const bf16_t* b1p = &BsT[wn + 16 + nc][0];
        FragU af0, af1, bf0, bf1;
#pragma unroll
        for (int v = 0; v < 8; ++v) {
            int kb  = (v < 4) ? (v * 2 + half * 8) : (16 + (v - 4) * 2 + half * 8);
            int kb2 = half * 16 + v * 2;
            af0.u[v] = *(const unsigned*)(a0p + kb);
            af1.u[v] = *(const unsigned*)(a1p + kb);
            bf0.u[v] = *(const unsigned*)(b0p + kb);
            bf1.u[v] = *(const unsigned*)(b1p + kb);
        }
        acc00 = __builtin_amdgcn_wmma_f32_16x16x32_bf16(false, af0.v, false, bf0.v,
                                                        (short)0, acc00, false, false);
        acc01 = __builtin_amdgcn_wmma_f32_16x16x32_bf16(false, af0.v, false, bf1.v,
                                                        (short)0, acc01, false, false);
        acc10 = __builtin_amdgcn_wmma_f32_16x16x32_bf16(false, af1.v, false, bf0.v,
                                                        (short)0, acc10, false, false);
        acc11 = __builtin_amdgcn_wmma_f32_16x16x32_bf16(false, af1.v, false, bf1.v,
                                                        (short)0, acc11, false, false);
    };

    const bool fastAll = (m0 + 128 <= M) && (n0 + 64 <= N) && ((Kdim & 31) == 0);

    if (fastAll) {
        // ---- software-pipelined wide path ----
        const int r0a = tid >> 3;          // A: 8 threads/row, 4 rows apart 32
        const int kca = (tid & 7) * 4;
        const int krb = tid >> 4;          // B transB=0: 16 threads/k-row
        const int ncb = (tid & 15) * 4;
        float4 ra[4], rb[2];

        auto loadTiles = [&](int k0) {
#pragma unroll
            for (int it = 0; it < 4; ++it)
                ra[it] = *(const float4*)(A + (long long)(m0 + r0a + it * 32) * lda + (k0 + kca));
            if (transB) {
#pragma unroll
                for (int it = 0; it < 2; ++it)
                    rb[it] = *(const float4*)(Bp + (long long)(n0 + r0a + it * 32) * ldb + (k0 + kca));
            } else {
#pragma unroll
                for (int it = 0; it < 2; ++it)
                    rb[it] = *(const float4*)(Bp + (long long)(k0 + krb + it * 16) * ldb + (n0 + ncb));
            }
        };
        auto storeTiles = [&]() {
#pragma unroll
            for (int it = 0; it < 4; ++it) {
                union { unsigned u[2]; unsigned long long ull; } w;
                w.u[0] = pack2(ra[it].x * scaleA, ra[it].y * scaleA);
                w.u[1] = pack2(ra[it].z * scaleA, ra[it].w * scaleA);
                *(unsigned long long*)&As[r0a + it * 32][kca] = w.ull;
            }
            if (transB) {
#pragma unroll
                for (int it = 0; it < 2; ++it) {
                    union { unsigned u[2]; unsigned long long ull; } w;
                    w.u[0] = pack2(rb[it].x, rb[it].y);
                    w.u[1] = pack2(rb[it].z, rb[it].w);
                    *(unsigned long long*)&BsT[r0a + it * 32][kca] = w.ull;
                }
            } else {
#pragma unroll
                for (int it = 0; it < 2; ++it) {
                    int k = krb + it * 16;
                    BsT[ncb + 0][k] = (bf16_t)rb[it].x;
                    BsT[ncb + 1][k] = (bf16_t)rb[it].y;
                    BsT[ncb + 2][k] = (bf16_t)rb[it].z;
                    BsT[ncb + 3][k] = (bf16_t)rb[it].w;
                }
            }
        };

        loadTiles(0);
        int k0 = 0;
        while (true) {
            storeTiles();
            __syncthreads();
            int kn = k0 + 32;
            if (kn < Kdim) loadTiles(kn);   // overlap next-tile loads with WMMA
            computeStep();
            __syncthreads();
            if (kn >= Kdim) break;
            k0 = kn;
        }
    } else {
        // ---- generic guarded path (edge tiles, K=16 GEMM) ----
        for (int k0 = 0; k0 < Kdim; k0 += 32) {
            for (int i = tid; i < 128 * 32; i += 256) {
                int r = i >> 5, c = i & 31;
                int gm = m0 + r, gk = k0 + c;
                float v = 0.f;
                if (gm < M && gk < Kdim) v = A[(long long)gm * lda + gk] * scaleA;
                As[r][c] = (bf16_t)v;
            }
            for (int i = tid; i < 32 * 64; i += 256) {
                int r = i >> 6, c = i & 63;      // r = k, c = n
                int gk = k0 + r, gn = n0 + c;
                float v = 0.f;
                if (gk < Kdim && gn < N)
                    v = transB ? Bp[(long long)gn * ldb + gk] : Bp[(long long)gk * ldb + gn];
                BsT[c][r] = (bf16_t)v;
            }
            __syncthreads();
            computeStep();
            __syncthreads();
        }
    }

    // ---- Write out. D layout: VGPR r -> M = r + 8*half, N = lane&15. ----
#pragma unroll
    for (int tm = 0; tm < 2; ++tm) {
#pragma unroll
        for (int tn = 0; tn < 2; ++tn) {
            const v8f* acc = (tm == 0) ? ((tn == 0) ? &acc00 : &acc01)
                                       : ((tn == 0) ? &acc10 : &acc11);
            int gn = n0 + wn + tn * 16 + nc;
            if (gn >= N) continue;
#pragma unroll
            for (int r = 0; r < 8; ++r) {
                int gm = m0 + wm + tm * 16 + half * 8 + r;
                if (gm >= M) continue;
                float v = (*acc)[r];
                if (biasMode == 1) v += bias[gn];
                else if (biasMode == 2) v += bias[gm];
                if (act == 1)      v = 0.5f * v * (1.0f + erff(v * 0.70710678118f));
                else if (act == 2) v = (v > 20.f) ? v : log1pf(expf(v));
                if (Res) v += Res[(long long)gm * ldrM + (long long)gn * ldrN];
                C[(long long)gm * ldc + gn] = v;
            }
        }
    }
}

// ---------------------------------------------------------------------------
// LayerNorm-and-mix: mix = ln(xn;g1,b1) + 0.1*ln(xo;g1o,b1o); lno = ln(xo).
// ---------------------------------------------------------------------------
__global__ __launch_bounds__(256)
void ln_mix_kernel(const float* __restrict__ xn, const float* __restrict__ xo,
                   const float* __restrict__ g1, const float* __restrict__ b1,
                   const float* __restrict__ g1o, const float* __restrict__ b1o,
                   float* __restrict__ mix, float* __restrict__ lno, int rows, int n)
{
    int wave = threadIdx.x >> 5, lane = threadIdx.x & 31;
    int r = blockIdx.x * 8 + wave;
    if (r >= rows) return;
    const float* p1 = xn + (long long)r * n;
    const float* p2 = xo + (long long)r * n;
    float s1 = 0.f, q1 = 0.f, s2 = 0.f, q2 = 0.f;
    for (int j = lane; j < n; j += 32) {
        float a = p1[j], c = p2[j];
        s1 += a; q1 += a * a; s2 += c; q2 += c * c;
    }
    for (int m = 16; m; m >>= 1) {
        s1 += __shfl_xor(s1, m, 32); q1 += __shfl_xor(q1, m, 32);
        s2 += __shfl_xor(s2, m, 32); q2 += __shfl_xor(q2, m, 32);
    }
    float inv = 1.f / (float)n;
    float m1 = s1 * inv, v1 = q1 * inv - m1 * m1, r1 = rsqrtf(v1 + 1e-5f);
    float m2 = s2 * inv, v2 = q2 * inv - m2 * m2, r2 = rsqrtf(v2 + 1e-5f);
    for (int j = lane; j < n; j += 32) {
        float a = (p1[j] - m1) * r1 * g1[j] + b1[j];
        float c = (p2[j] - m2) * r2 * g1o[j] + b1o[j];
        mix[(long long)r * n + j] = a + 0.1f * c;
        lno[(long long)r * n + j] = c;
    }
}

// ---------------------------------------------------------------------------
// Generic strided LayerNorm. One wave per row.
// ---------------------------------------------------------------------------
__global__ __launch_bounds__(256)
void ln_kernel(const float* __restrict__ in, float* __restrict__ out,
               const float* __restrict__ g, const float* __restrict__ bb,
               int rows, int n, int innerRows,
               long long rowOuter, long long rowInner, long long estride,
               long long oOuter, long long oInner, long long oEstride)
{
    int wave = threadIdx.x >> 5, lane = threadIdx.x & 31;
    int r = blockIdx.x * 8 + wave;
    if (r >= rows) return;
    long long base  = (long long)(r / innerRows) * rowOuter + (long long)(r % innerRows) * rowInner;
    long long obase = (long long)(r / innerRows) * oOuter  + (long long)(r % innerRows) * oInner;
    float s = 0.f, q = 0.f;
    for (int j = lane; j < n; j += 32) {
        float x = in[base + j * estride];
        s += x; q += x * x;
    }
    for (int m = 16; m; m >>= 1) { s += __shfl_xor(s, m, 32); q += __shfl_xor(q, m, 32); }
    float mean = s / (float)n;
    float var  = q / (float)n - mean * mean;
    float rstd = rsqrtf(var + 1e-5f);
    for (int j = lane; j < n; j += 32) {
        float x = in[base + j * estride];
        out[obase + j * oEstride] = (x - mean) * rstd * g[j] + bb[j];
    }
}

// ---------------------------------------------------------------------------
// Row softmax for n==256.
// ---------------------------------------------------------------------------
__global__ __launch_bounds__(256)
void softmax256_kernel(float* __restrict__ data, int rows)
{
    int wave = threadIdx.x >> 5, lane = threadIdx.x & 31;
    int r = blockIdx.x * 8 + wave;
    if (r >= rows) return;
    float* p = data + (long long)r * 256;
    float v[8];
    float mx = -3.4e38f;
#pragma unroll
    for (int i = 0; i < 8; ++i) { v[i] = p[lane + i * 32]; mx = fmaxf(mx, v[i]); }
    for (int m = 16; m; m >>= 1) mx = fmaxf(mx, __shfl_xor(mx, m, 32));
    float sum = 0.f;
#pragma unroll
    for (int i = 0; i < 8; ++i) { v[i] = __expf(v[i] - mx); sum += v[i]; }
    for (int m = 16; m; m >>= 1) sum += __shfl_xor(sum, m, 32);
    float inv = 1.f / sum;
#pragma unroll
    for (int i = 0; i < 8; ++i) p[lane + i * 32] = v[i] * inv;
}

// ---------------------------------------------------------------------------
// Causal depthwise conv (taps=4) over L + SiLU.
// ---------------------------------------------------------------------------
__global__ __launch_bounds__(256)
void conv_silu_kernel(const float* __restrict__ xz, const float* __restrict__ cw,
                      const float* __restrict__ cb, float* __restrict__ u)
{
    long long idx = (long long)blockIdx.x * 256 + threadIdx.x;
    long long total = (long long)BB * LL * DIN;
    if (idx >= total) return;
    int d = (int)(idx % DIN);
    long long bl = idx / DIN;
    int l = (int)(bl % LL);
    int b = (int)(bl / LL);
    const float* base = xz + (long long)b * LL * 1024 + d;
    float acc = cb[d];
#pragma unroll
    for (int k = 0; k < 4; ++k) {
        int ls = l - 3 + k;
        if (ls >= 0) acc += cw[d * 4 + k] * base[(long long)ls * 1024];
    }
    u[idx] = acc / (1.f + __expf(-acc));
}

// ---------------------------------------------------------------------------
// Selective scan: thread per (b,d), 64-wide state in VGPRs. B_t||C_t is 128
// contiguous floats (dbl row elements 16..143) -> staged into LDS each step
// with GLOBAL_LOAD_ASYNC_TO_LDS_B128 (one b128 per lane of wave 0 = 512 B),
// completed with s_wait_asynccnt. Fused u*D and SiLU(z) gating.
// ---------------------------------------------------------------------------
__global__ __launch_bounds__(256)
void scan_kernel(const float* __restrict__ u, const float* __restrict__ dt,
                 const float* __restrict__ dbl, const float* __restrict__ xz,
                 const float* __restrict__ Alog, const float* __restrict__ Dp,
                 float* __restrict__ y)
{
    int b = blockIdx.x >> 1;
    int d = ((blockIdx.x & 1) << 8) + threadIdx.x;
    float h[DSN], Ad[DSN];
#pragma unroll
    for (int s = 0; s < DSN; ++s) {
        h[s] = 0.f;
        Ad[s] = -__expf(Alog[d * DSN + s]);
    }
    float Dd = Dp[d];
    __shared__ __align__(16) float sBC[128];   // [0..63] = B_t, [64..127] = C_t
    for (int t = 0; t < LL; ++t) {
        long long row = (long long)b * LL + t;
        __syncthreads();
#ifdef HAVE_ASYNC_LDS
        if (threadIdx.x < 32) {
            __builtin_amdgcn_global_load_async_to_lds_b128(
                (g_v4i_ptr)(dbl + row * 144 + 16 + threadIdx.x * 4),
                (l_v4i_ptr)(&sBC[threadIdx.x * 4]),
                0, 0);
#if __has_builtin(__builtin_amdgcn_s_wait_asynccnt)
            __builtin_amdgcn_s_wait_asynccnt(0);
#else
            asm volatile("s_wait_asynccnt 0" ::: "memory");
#endif
        }
#else
        if (threadIdx.x < 128) sBC[threadIdx.x] = dbl[row * 144 + 16 + threadIdx.x];
#endif
        __syncthreads();
        float ut  = u[row * DIN + d];
        float dtt = dt[row * DIN + d];
        float du  = dtt * ut;
        float acc = 0.f;
#pragma unroll
        for (int s = 0; s < DSN; ++s) {
            float dA = __expf(dtt * Ad[s]);
            h[s] = dA * h[s] + du * sBC[s];
            acc += h[s] * sBC[64 + s];
        }
        acc += ut * Dd;
        float z = xz[row * 1024 + DIN + d];
        acc *= z / (1.f + __expf(-z));
        y[row * DIN + d] = acc;
    }
}

// ---------------------------------------------------------------------------
// xs3[b,c,p] = xs2[b,c,p] + mamO[b,p,c]  (LDS-tiled transpose-add, 32x32)
// ---------------------------------------------------------------------------
__global__ __launch_bounds__(256)
void add_transpose_kernel(const float* __restrict__ xs2, const float* __restrict__ mamO,
                          float* __restrict__ xs3)
{
    __shared__ float tile[32][33];
    int b  = blockIdx.z;
    int c0 = blockIdx.y * 32;
    int p0 = blockIdx.x * 32;
    int lane = threadIdx.x & 31;
    int w    = threadIdx.x >> 5;
    long long base = (long long)b * (C4N * PP);
#pragma unroll
    for (int it = 0; it < 4; ++it) {
        int i = w + it * 8;   // p-row within tile; lane runs over c (coalesced)
        tile[i][lane] = mamO[base + (long long)(p0 + i) * C4N + (c0 + lane)];
    }
    __syncthreads();
#pragma unroll
    for (int it = 0; it < 4; ++it) {
        int i = w + it * 8;   // c-row within tile; lane runs over p (coalesced)
        long long o = base + (long long)(c0 + i) * PP + (p0 + lane);
        xs3[o] = xs2[o] + tile[lane][i];
    }
}

// ---------------------------------------------------------------------------
// Host-side launch helper
// ---------------------------------------------------------------------------
static inline void gemm(hipStream_t st,
    int M, int N, int K, int nOuter, int nInner,
    const float* A, long long sAo, long long sAi, int lda, float scaleA,
    const float* Bp, long long sBo, long long sBi, int ldb, int transB,
    float* C, long long sCo, long long sCi, int ldc,
    const float* bias = nullptr, int biasMode = 0,
    const float* Res = nullptr, long long sRo = 0, long long sRi = 0,
    long long ldrM = 0, long long ldrN = 1,
    int act = 0)
{
    dim3 grid((N + 63) / 64, (M + 127) / 128, nOuter * nInner);
    wmma_gemm_kernel<<<grid, 256, 0, st>>>(
        A, sAo, sAi, lda, scaleA,
        Bp, sBo, sBi, ldb, transB,
        C, sCo, sCi, ldc,
        bias, biasMode,
        Res, sRo, sRi, ldrM, ldrN,
        M, N, K, nInner, act);
}

extern "C" void kernel_launch(void* const* d_in, const int* in_sizes, int n_in,
                              void* d_out, int out_size, void* d_ws, size_t ws_size,
                              hipStream_t stream)
{
    (void)in_sizes; (void)n_in; (void)out_size; (void)ws_size;
    const float* x        = (const float*)d_in[0];
    const float* x_ori    = (const float*)d_in[1];
    const float* w_narrow = (const float*)d_in[3];
    const float* b_narrow = (const float*)d_in[4];
    const float* w_broad  = (const float*)d_in[5];
    const float* b_broad  = (const float*)d_in[6];
    const float* ln1_g    = (const float*)d_in[7];
    const float* ln1_b    = (const float*)d_in[8];
    const float* ln1o_g   = (const float*)d_in[9];
    const float* ln1o_b   = (const float*)d_in[10];
    const float* w_qkv    = (const float*)d_in[11];
    const float* b_qkv    = (const float*)d_in[12];
    const float* w_out    = (const float*)d_in[13];
    const float* b_out    = (const float*)d_in[14];
    const float* mn1_g    = (const float*)d_in[15];
    const float* mn1_b    = (const float*)d_in[16];
    const float* m_in_w   = (const float*)d_in[17];
    const float* m_conv_w = (const float*)d_in[18];
    const float* m_conv_b = (const float*)d_in[19];
    const float* m_xproj  = (const float*)d_in[20];
    const float* m_dt_w   = (const float*)d_in[21];
    const float* m_dt_b   = (const float*)d_in[22];
    const float* m_Alog   = (const float*)d_in[23];
    const float* m_D      = (const float*)d_in[24];
    const float* m_out_w  = (const float*)d_in[25];
    const float* mn2_g    = (const float*)d_in[26];
    const float* mn2_b    = (const float*)d_in[27];
    const float* f1w      = (const float*)d_in[28];
    const float* f1b      = (const float*)d_in[29];
    const float* f2w      = (const float*)d_in[30];
    const float* f2b      = (const float*)d_in[31];
    float* out = (float*)d_out;
    float* ws  = (float*)d_ws;

    const long long U  = (long long)BB * C4N * PP;     // 4,718,592 floats
    const long long PB = (long long)C4N * PP;          // 147,456 per-batch
    const long long XB = (long long)CHN * PP;          // 589,824 per-batch

    float* xn    = ws + 0 * U;
    float* xo    = ws + 1 * U;
    float* xsmix = ws + 2 * U;
    float* xosln = ws + 3 * U;
    float* Qb    = ws + 4 * U;
    float* Kb    = ws + 5 * U;
    float* Vb    = ws + 6 * U;
    float* scor  = ws + 7 * U;          // 4U
    float* attnO = ws + 11 * U;
    float* xs2   = ws + 12 * U;
    float* dtbuf = ws + 0 * U;          // reuse xn,xo (2U)
    float* mln   = ws + 3 * U;          // reuse xosln
    float* xz    = ws + 7 * U;          // reuse scores (4U)
    float* ubuf  = ws + 4 * U;          // reuse Q,K (2U)
    float* dblb  = ws + 6 * U;          // reuse V
    float* ysc   = ws + 13 * U;         // 2U fresh
    float* rbuf  = ws + 2 * U;          // reuse xsmix
    float* ln2b  = ws + 4 * U;          // reuse u (dead after scan)
    float* hfb   = ws + 5 * U;
    float* mamO  = ws + 11 * U;         // reuse attnO
    float* xs3   = ws + 4 * U;          // reuse ln2b (dead after ffn1)

    // 1-2: conv1x1 narrow
    gemm(stream, C4N, PP, CHN, BB, 1,
         w_narrow, 0, 0, CHN, 1.f,
         x, XB, 0, PP, 0,
         xn, PB, 0, PP, b_narrow, 2);
    gemm(stream, C4N, PP, CHN, BB, 1,
         w_narrow, 0, 0, CHN, 1.f,
         x_ori, XB, 0, PP, 0,
         xo, PB, 0, PP, b_narrow, 2);

    // 3: ln + mix
    ln_mix_kernel<<<(BB * C4N + 7) / 8, 256, 0, stream>>>(
        xn, xo, ln1_g, ln1_b, ln1o_g, ln1o_b, xsmix, xosln, BB * C4N, PP);

    // 4-6: Q,K,V projections
    const long long EE2 = (long long)EE * EE;
    gemm(stream, SS, EE, EE, BB, 1, xsmix, PB, 0, EE, 1.f,
         w_qkv + 0 * EE2, 0, 0, EE, 1, Qb, PB, 0, EE, b_qkv + 0 * EE, 1);
    gemm(stream, SS, EE, EE, BB, 1, xosln, PB, 0, EE, 1.f,
         w_qkv + 1 * EE2, 0, 0, EE, 1, Kb, PB, 0, EE, b_qkv + 1 * EE, 1);
    gemm(stream, SS, EE, EE, BB, 1, xosln, PB, 0, EE, 1.f,
         w_qkv + 2 * EE2, 0, 0, EE, 1, Vb, PB, 0, EE, b_qkv + 2 * EE, 1);

    // 7: scores[b,h] = (Q_h / 8) @ K_h^T  (headed batching, nInner=9)
    gemm(stream, SS, SS, HD, BB, NHD,
         Qb, PB, HD, EE, 0.125f,
         Kb, PB, HD, EE, 1,
         scor, (long long)NHD * SS * SS, (long long)SS * SS, SS);

    // 8: softmax
    softmax256_kernel<<<(BB * NHD * SS + 7) / 8, 256, 0, stream>>>(scor, BB * NHD * SS);

    // 9: AV -> attnO (B, S, E)
    gemm(stream, SS, HD, SS, BB, NHD,
         scor, (long long)NHD * SS * SS, (long long)SS * SS, SS, 1.f,
         Vb, PB, HD, EE, 0,
         attnO, PB, HD, EE);

    // 10: out-proj + residual(xs_mix) -> xs2
    gemm(stream, SS, EE, EE, BB, 1, attnO, PB, 0, EE, 1.f,
         w_out, 0, 0, EE, 1, xs2, PB, 0, EE, b_out, 1,
         xsmix, PB, 0, EE, 1, 0);

    // 11: mamba ln1 on xt = xs2^T (strided read), dense write (B, L, C)
    ln_kernel<<<(BB * LL + 7) / 8, 256, 0, stream>>>(
        xs2, mln, mn1_g, mn1_b, BB * LL, C4N, LL,
        PB, 1, PP, PB, C4N, 1);

    // 12: xz = mln @ in_w^T
    gemm(stream, LL, 2 * DIN, C4N, BB, 1, mln, PB, 0, C4N, 1.f,
         m_in_w, 0, 0, C4N, 1, xz, (long long)LL * 2 * DIN, 0, 2 * DIN);

    // 13: causal conv + SiLU -> u
    {
        long long total = (long long)BB * LL * DIN;
        conv_silu_kernel<<<(unsigned)((total + 255) / 256), 256, 0, stream>>>(
            xz, m_conv_w, m_conv_b, ubuf);
    }

    // 14: dbl = u @ xproj^T
    gemm(stream, LL, RN + 2 * DSN, DIN, BB, 1, ubuf, (long long)LL * DIN, 0, DIN, 1.f,
         m_xproj, 0, 0, DIN, 1, dblb, (long long)LL * 144, 0, 144);

    // 15: dt = softplus(dbl[:, :16] @ dt_w^T + dt_b)
    gemm(stream, LL, DIN, RN, BB, 1, dblb, (long long)LL * 144, 0, 144, 1.f,
         m_dt_w, 0, 0, RN, 1, dtbuf, (long long)LL * DIN, 0, DIN, m_dt_b, 1,
         nullptr, 0, 0, 0, 1, 2);

    // 16: selective scan
    scan_kernel<<<BB * (DIN / 256), 256, 0, stream>>>(
        ubuf, dtbuf, dblb, xz, m_Alog, m_D, ysc);

    // 17: r = xt + y @ out_w^T (residual is strided transpose of xs2)
    gemm(stream, LL, C4N, DIN, BB, 1, ysc, (long long)LL * DIN, 0, DIN, 1.f,
         m_out_w, 0, 0, DIN, 1, rbuf, PB, 0, C4N, nullptr, 0,
         xs2, PB, 0, 1, PP, 0);

    // 18: ln2 on r
    ln_kernel<<<(BB * LL + 7) / 8, 256, 0, stream>>>(
        rbuf, ln2b, mn2_g, mn2_b, BB * LL, C4N, LL,
        PB, C4N, 1, PB, C4N, 1);

    // 19: hf = gelu(ln2 @ f1^T + b1)
    gemm(stream, LL, C4N, C4N, BB, 1, ln2b, PB, 0, C4N, 1.f,
         f1w, 0, 0, C4N, 1, hfb, PB, 0, C4N, f1b, 1,
         nullptr, 0, 0, 0, 1, 1);

    // 20: mamO = hf @ f2^T + b2
    gemm(stream, LL, C4N, C4N, BB, 1, hfb, PB, 0, C4N, 1.f,
         f2w, 0, 0, C4N, 1, mamO, PB, 0, C4N, f2b, 1);

    // 20b: xs3 = xs2 + mamO^T (tiled transpose-add; frees broad conv for the
    //      pipelined wide GEMM path)
    add_transpose_kernel<<<dim3(PP / 32, C4N / 32, BB), 256, 0, stream>>>(xs2, mamO, xs3);

    // 21: final broad conv
    gemm(stream, CHN, PP, C4N, BB, 1,
         w_broad, 0, 0, C4N, 1.f,
         xs3, PB, 0, PP, 0,
         out, XB, 0, PP, b_broad, 2);
}
